// MultiHeadAttention_84404697301527
// MI455X (gfx1250) — matrix operand order repaired and checked by reading, not compiled, not run on hardware
//
#include <hip/hip_runtime.h>
#include <cstdint>

// ---------------------------------------------------------------------------
// MHA forward for MI455X (gfx1250, wave32): WMMA bf16 16x16x32 f32-accum,
// TDM (tensor_load_to_lds) staging with native LDS row padding, double-
// buffered K tiles in the projection GEMMs, causal flash attention.
// B=2, T=2048, DIM=1024, H=16, Dh=64
// ---------------------------------------------------------------------------

typedef __attribute__((ext_vector_type(16))) __bf16        v16bf;
typedef __attribute__((ext_vector_type(8)))  float         v8f;
typedef __attribute__((ext_vector_type(4)))  unsigned int  u32x4;
typedef __attribute__((ext_vector_type(8)))  int           i32x8;
typedef __attribute__((ext_vector_type(4)))  int           i32x4;

union FragAB { v16bf v; uint4 u[2]; };
union Pack8  { uint4 q; unsigned short s[8]; };

static constexpr int kB    = 2;
static constexpr int kT    = 2048;
static constexpr int kDIM  = 1024;
static constexpr int kH    = 16;
static constexpr int kDh   = 64;
static constexpr int kMtok = kB * kT;          // 4096

#if defined(__has_builtin)
#  if __has_builtin(__builtin_amdgcn_tensor_load_to_lds) && \
      __has_builtin(__builtin_amdgcn_s_wait_tensorcnt)
#    define HAVE_TDM 1
#  endif
#endif
#ifndef HAVE_TDM
#  define HAVE_TDM 0
#endif

static __device__ __forceinline__ unsigned short f2bf(float f) {
  unsigned int u = __float_as_uint(f);
  u += 0x7FFFu + ((u >> 16) & 1u);             // round-to-nearest-even
  return (unsigned short)(u >> 16);
}
static __device__ __forceinline__ float bf2f(unsigned short s) {
  return __uint_as_float(((unsigned int)s) << 16);
}
static __device__ __forceinline__ v8f v8f_zero() {
  v8f z;
  #pragma unroll
  for (int e = 0; e < 8; ++e) z[e] = 0.0f;
  return z;
}

#if HAVE_TDM
// ---------------------------------------------------------------------------
// TDM: async 2D bf16 tile load Global->LDS with LDS row padding.
// D# layout per CDNA5 ISA. dims/strides in
// elements (data_size=2B); global_addr in bytes; lds_addr = LDS byte offset.
// pad_i: 3 => pad after every 16 DWORDs; 4 => after every 32 DWORDs.
// pad_a: 3 => pad 4 DWORDs (16B).
// ---------------------------------------------------------------------------
static __device__ __forceinline__ void tdm_load_2d_bf16(
    unsigned lds_off, const void* gptr,
    unsigned tile_d0, unsigned tile_d1,
    unsigned tensor_d0, unsigned tensor_d1, unsigned row_stride,
    unsigned pad_i, unsigned pad_a)
{
  unsigned long long ga = (unsigned long long)(uintptr_t)gptr;
  u32x4 g0;
  g0[0] = 1u;                                        // count=1, user mode
  g0[1] = lds_off;                                   // lds_addr
  g0[2] = (unsigned)(ga & 0xFFFFFFFFu);              // global_addr lo
  g0[3] = (unsigned)((ga >> 32) & 0x01FFFFFFu)       // global_addr hi (57b)
          | (2u << 30);                              // type = 2 ("image")
  i32x8 g1;
  g1[0] = (int)((1u << 16)                           // data_size = 2 bytes
          | (1u << 20)                               // pad_enable
          | (pad_i << 22) | (pad_a << 25));
  g1[1] = (int)((tensor_d0 & 0xFFFFu) << 16);        // tensor_dim0[15:0]
  g1[2] = (int)((tensor_d0 >> 16) | (tensor_d1 << 16));
  g1[3] = (int)((tensor_d1 >> 16) | (tile_d0 << 16));
  g1[4] = (int)(tile_d1 & 0xFFFFu);                  // tile_dim1 (tile_dim2=0)
  g1[5] = (int)row_stride;                           // tensor_dim0_stride lo
  g1[6] = 0;
  g1[7] = 0;
  i32x4 z4; z4[0] = z4[1] = z4[2] = z4[3] = 0;
#if defined(__clang_major__) && (__clang_major__ >= 23)
  i32x8 z8;
  #pragma unroll
  for (int e = 0; e < 8; ++e) z8[e] = 0;
  __builtin_amdgcn_tensor_load_to_lds(g0, g1, z4, z4, z8, 0);
#else
  __builtin_amdgcn_tensor_load_to_lds(g0, g1, z4, z4, 0);
#endif
}
static __device__ __forceinline__ unsigned lds_off_of(const void* p) {
  return (unsigned)(uintptr_t)p;                     // low 32 bits = LDS offset
}
#endif  // HAVE_TDM

// ---------------------------------------------------------------------------
// Kernel 1: fp32 -> bf16 cast (grid-stride)
// ---------------------------------------------------------------------------
__global__ void cast_f32_bf16(const float* __restrict__ src,
                              unsigned short* __restrict__ dst, int n) {
  for (int i = blockIdx.x * blockDim.x + threadIdx.x; i < n;
       i += gridDim.x * blockDim.x)
    dst[i] = f2bf(src[i]);
}

// ---------------------------------------------------------------------------
// Kernel 2: C[m,n] = sum_k A[m,k] * W[n,k]   (A: MxK bf16, W: NxK bf16)
// 128x128 macro tile, 4 waves each owning 64x64 (16 WMMA per K-step),
// double-buffered TDM staging of the 128x32 K-tiles.
// ---------------------------------------------------------------------------
static constexpr int TM = 128, TN = 128, TK = 32;
static constexpr int LSTR = 40;                 // 32 + 8 pad (bf16 elems)

template <bool OUT_BF16>
__global__ __launch_bounds__(128) void gemm_xwT(
    const unsigned short* __restrict__ A,
    const unsigned short* __restrict__ W,
    void* __restrict__ Cout, int M, int N, int K)
{
  __shared__ unsigned short As[2][TM * LSTR];
  __shared__ unsigned short Bs[2][TN * LSTR];

  const int tid  = threadIdx.x;
  const int wave = tid >> 5;                    // 0..3
  const int lane = tid & 31;
  const int l15  = lane & 15;
  const int half = lane >> 4;

  const int m0 = blockIdx.y * TM;
  const int n0 = blockIdx.x * TN;
  const int wm = (wave >> 1) * 64;              // 4 m-subtiles
  const int wn = (wave & 1) * 64;               // 4 n-subtiles

  v8f acc[4][4];
  #pragma unroll
  for (int i = 0; i < 4; ++i)
    #pragma unroll
    for (int j = 0; j < 4; ++j) acc[i][j] = v8f_zero();

  const int kbA = half * 8;                     // A frag K base (16-bit layout)
  const int kbB = half * 8;                     // B frag K base

#if HAVE_TDM
  if (wave == 0) {                              // preload buffer 0
    tdm_load_2d_bf16(lds_off_of(&As[0][0]), &A[(size_t)m0 * K],
                     TK, TM, (unsigned)K, (unsigned)M, (unsigned)K, 3u, 3u);
    tdm_load_2d_bf16(lds_off_of(&Bs[0][0]), &W[(size_t)n0 * K],
                     TK, TN, (unsigned)K, (unsigned)N, (unsigned)K, 3u, 3u);
  }
#endif

  for (int k0 = 0; k0 < K; k0 += TK) {
    const int cur = (k0 >> 5) & 1;
    __syncthreads();                            // prior reads of buf cur^1 done
#if HAVE_TDM
    if (wave == 0) {
      if (k0 + TK < K) {
        tdm_load_2d_bf16(lds_off_of(&As[cur ^ 1][0]),
                         &A[(size_t)m0 * K + k0 + TK],
                         TK, TM, (unsigned)K, (unsigned)M, (unsigned)K, 3u, 3u);
        tdm_load_2d_bf16(lds_off_of(&Bs[cur ^ 1][0]),
                         &W[(size_t)n0 * K + k0 + TK],
                         TK, TN, (unsigned)K, (unsigned)N, (unsigned)K, 3u, 3u);
        __builtin_amdgcn_s_wait_tensorcnt(2);   // current buffer complete
      } else {
        __builtin_amdgcn_s_wait_tensorcnt(0);
      }
    }
#else
    // manual staging fallback: 128 threads x 4 uint4 per tile
    #pragma unroll
    for (int it = 0; it < 4; ++it) {
      int i = tid + it * 128, row = i >> 2, col8 = (i & 3) * 8;
      *(uint4*)&As[cur][row * LSTR + col8] =
          *(const uint4*)&A[(size_t)(m0 + row) * K + k0 + col8];
      *(uint4*)&Bs[cur][row * LSTR + col8] =
          *(const uint4*)&W[(size_t)(n0 + row) * K + k0 + col8];
    }
    if (k0 + TK < K) {
      __builtin_prefetch(&A[(size_t)(m0 + lane) * K + k0 + TK], 0, 1);
      __builtin_prefetch(&W[(size_t)(n0 + lane) * K + k0 + TK], 0, 1);
    }
#endif
    __syncthreads();

    FragAB af[4], bfr[4];
    #pragma unroll
    for (int i = 0; i < 4; ++i) {
      const unsigned short* p = &As[cur][(wm + i * 16 + l15) * LSTR];
      af[i].u[0] = *(const uint4*)(p + kbA);
      af[i].u[1] = *(const uint4*)(p + kbA + 16);
    }
    #pragma unroll
    for (int j = 0; j < 4; ++j) {
      const unsigned short* p = &Bs[cur][(wn + j * 16 + l15) * LSTR];
      bfr[j].u[0] = *(const uint4*)(p + kbB);
      bfr[j].u[1] = *(const uint4*)(p + kbB + 16);
    }
    #pragma unroll
    for (int i = 0; i < 4; ++i)
      #pragma unroll
      for (int j = 0; j < 4; ++j)
        acc[i][j] = __builtin_amdgcn_wmma_f32_16x16x32_bf16(
            false, af[i].v, false, bfr[j].v, (short)0, acc[i][j], false, false);
  }

  // epilogue: C/D layout -> row = r + 8*half, col = l15 (per 16x16 subtile)
  #pragma unroll
  for (int i = 0; i < 4; ++i) {
    int rowb = m0 + wm + i * 16 + 8 * half;
    #pragma unroll
    for (int j = 0; j < 4; ++j) {
      int col = n0 + wn + j * 16 + l15;
      #pragma unroll
      for (int r = 0; r < 8; ++r) {
        float val = acc[i][j][r];
        size_t idx = (size_t)(rowb + r) * N + col;
        if (OUT_BF16) ((unsigned short*)Cout)[idx] = f2bf(val);
        else          ((float*)Cout)[idx] = val;
      }
    }
  }
}

// ---------------------------------------------------------------------------
// Kernel 3: RoPE on q,k + repack q,k,v from [B,T,H*Dh] to [B,H,T,Dh].
// Softmax scale Dh^-0.5 = 0.125 is folded into q here.
// ---------------------------------------------------------------------------
__global__ __launch_bounds__(64) void rope_pack(
    const unsigned short* __restrict__ qp, const unsigned short* __restrict__ kp,
    const unsigned short* __restrict__ vp,
    unsigned short* __restrict__ qh, unsigned short* __restrict__ kh,
    unsigned short* __restrict__ vh)
{
  const int d   = threadIdx.x;                  // 0..63
  const int idx = blockIdx.x;                   // b*T*H + t*H + h
  const int h = idx % kH;
  const int t = (idx / kH) % kT;
  const int b = idx / (kH * kT);

  const size_t src = ((size_t)(b * kT + t)) * kDIM + h * kDh + d;
  const size_t dst = (((size_t)(b * kH + h)) * kT + t) * kDh + d;

  const int   i    = d & 31;
  const float freq = __expf(-(float)i * 0.28782313662425572f); // ln(1e4)/32
  const float ang  = (float)t * freq;
  const float c = __cosf(ang), s = __sinf(ang);

  const int   dpart = (d < 32) ? (d + 32) : (d - 32);
  const float sgn   = (d < 32) ? -1.0f : 1.0f;
  const size_t src2 = src - d + dpart;

  float q = bf2f(qp[src]), k = bf2f(kp[src]);
  float q2 = bf2f(qp[src2]), k2 = bf2f(kp[src2]);
  qh[dst] = f2bf((q * c + sgn * q2 * s) * 0.125f);   // pre-scaled Q
  kh[dst] = f2bf(k * c + sgn * k2 * s);
  vh[dst] = vp[src];
}

// ---------------------------------------------------------------------------
// Kernel 4: causal flash attention, one (b,h) x 64-query tile per workgroup.
// 4 waves; wave w owns queries [qbase + 16w, +16). Streams 64-key chunks.
// Q and K tiles staged by TDM; V transposed manually (TDM can't transpose).
// ---------------------------------------------------------------------------
static constexpr int FSTR = 72;                 // 64 + 8 pad

__global__ __launch_bounds__(128) void flash_attn(
    const unsigned short* __restrict__ Qh, const unsigned short* __restrict__ Kh,
    const unsigned short* __restrict__ Vh, unsigned short* __restrict__ Oattn)
{
  __shared__ unsigned short Qs[64 * FSTR];      // [q][dh]
  __shared__ unsigned short Ks[64 * FSTR];      // [key][dh]
  __shared__ unsigned short Vs[64 * FSTR];      // [dh][key]  (transposed)
  __shared__ unsigned short Ps[4][16 * FSTR];   // per-wave P scratch [q][key]

  const int tid  = threadIdx.x;
  const int wave = tid >> 5;
  const int lane = tid & 31;
  const int l15  = lane & 15;
  const int half = lane >> 4;
  const int kbA  = half * 8;
  const int kbB  = half * 8;

  const int bh    = blockIdx.y;                 // b*H + h
  const int qbase = blockIdx.x * 64;
  const unsigned short* Qp = Qh + (size_t)bh * kT * kDh;
  const unsigned short* Kp = Kh + (size_t)bh * kT * kDh;
  const unsigned short* Vp = Vh + (size_t)bh * kT * kDh;

#if HAVE_TDM
  if (wave == 0)
    tdm_load_2d_bf16(lds_off_of(&Qs[0]), &Qp[(size_t)qbase * kDh],
                     64u, 64u, 64u, (unsigned)kT, 64u, 4u, 3u);
#else
  #pragma unroll
  for (int it = 0; it < 4; ++it) {
    int i = tid + it * 128, row = i >> 3, col8 = (i & 7) * 8;
    *(uint4*)&Qs[row * FSTR + col8] =
        *(const uint4*)&Qp[(size_t)(qbase + row) * kDh + col8];
  }
#endif

  v8f o[4];
  #pragma unroll
  for (int j = 0; j < 4; ++j) o[j] = v8f_zero();
  float m_r[8], l_r[8];
  #pragma unroll
  for (int r = 0; r < 8; ++r) { m_r[r] = -3.0e38f; l_r[r] = 0.0f; }

  const int nchunks = blockIdx.x + 1;           // causal: keys up to query tile
  for (int cidx = 0; cidx < nchunks; ++cidx) {
    const int kc = cidx * 64;
    __syncthreads();
#if HAVE_TDM
    if (wave == 0)                              // async K chunk (overlaps V)
      tdm_load_2d_bf16(lds_off_of(&Ks[0]), &Kp[(size_t)kc * kDh],
                       64u, 64u, 64u, (unsigned)kT, 64u, 4u, 3u);
#endif
    // stage V transposed (all threads)
    #pragma unroll
    for (int it = 0; it < 4; ++it) {
      int i = tid + it * 128, row = i >> 3, col8 = (i & 7) * 8;
#if !HAVE_TDM
      *(uint4*)&Ks[row * FSTR + col8] =
          *(const uint4*)&Kp[(size_t)(kc + row) * kDh + col8];
#endif
      Pack8 vv; vv.q = *(const uint4*)&Vp[(size_t)(kc + row) * kDh + col8];
      #pragma unroll
      for (int e = 0; e < 8; ++e) Vs[(col8 + e) * FSTR + row] = vv.s[e];
    }
#if HAVE_TDM
    if (wave == 0) __builtin_amdgcn_s_wait_tensorcnt(0);
#endif
    __syncthreads();

    // S = Q K^T  (4 key-subtiles x 2 k-steps over Dh); Q pre-scaled by 0.125
    v8f s[4];
    #pragma unroll
    for (int n = 0; n < 4; ++n) s[n] = v8f_zero();
    #pragma unroll
    for (int ks = 0; ks < 2; ++ks) {
      FragAB qa;
      const unsigned short* p = &Qs[(wave * 16 + l15) * FSTR + ks * 32];
      qa.u[0] = *(const uint4*)(p + kbA);
      qa.u[1] = *(const uint4*)(p + kbA + 16);
      #pragma unroll
      for (int n = 0; n < 4; ++n) {
        FragAB kf;
        const unsigned short* pk = &Ks[(n * 16 + l15) * FSTR + ks * 32];
        kf.u[0] = *(const uint4*)(pk + kbB);
        kf.u[1] = *(const uint4*)(pk + kbB + 16);
        s[n] = __builtin_amdgcn_wmma_f32_16x16x32_bf16(
            false, qa.v, false, kf.v, (short)0, s[n], false, false);
      }
    }

    // causal mask
    const int qrb = qbase + wave * 16 + 8 * half;
    #pragma unroll
    for (int n = 0; n < 4; ++n) {
      const int col = kc + n * 16 + l15;
      #pragma unroll
      for (int r = 0; r < 8; ++r)
        if (col > qrb + r) s[n][r] = -3.0e38f;
    }

    // online softmax per row (stats shared by the 16 lanes of each half)
    #pragma unroll
    for (int r = 0; r < 8; ++r) {
      float mx = fmaxf(fmaxf(s[0][r], s[1][r]), fmaxf(s[2][r], s[3][r]));
      #pragma unroll
      for (int off = 1; off < 16; off <<= 1)
        mx = fmaxf(mx, __shfl_xor(mx, off, 16));
      float mnew = fmaxf(m_r[r], mx);
      float corr = __expf(m_r[r] - mnew);
      m_r[r] = mnew;
      float rs = 0.0f;
      #pragma unroll
      for (int n = 0; n < 4; ++n) {
        float pv = __expf(s[n][r] - mnew);
        s[n][r] = pv;
        rs += pv;
      }
      #pragma unroll
      for (int off = 1; off < 16; off <<= 1) rs += __shfl_xor(rs, off, 16);
      l_r[r] = l_r[r] * corr + rs;
      #pragma unroll
      for (int j = 0; j < 4; ++j) o[j][r] *= corr;
    }

    // spill P (C/D layout) -> per-wave LDS [q][key] for A-fragment reload
    #pragma unroll
    for (int n = 0; n < 4; ++n)
      #pragma unroll
      for (int r = 0; r < 8; ++r)
        Ps[wave][(r + 8 * half) * FSTR + n * 16 + l15] = f2bf(s[n][r]);
    __syncthreads();

    // O += P V  (2 k-steps over 64 keys x 4 dh-subtiles)
    #pragma unroll
    for (int ks = 0; ks < 2; ++ks) {
      FragAB pf;
      const unsigned short* pp = &Ps[wave][l15 * FSTR + ks * 32];
      pf.u[0] = *(const uint4*)(pp + kbA);
      pf.u[1] = *(const uint4*)(pp + kbA + 16);
      #pragma unroll
      for (int j = 0; j < 4; ++j) {
        FragAB vf;
        const unsigned short* pv = &Vs[(j * 16 + l15) * FSTR + ks * 32];
        vf.u[0] = *(const uint4*)(pv + kbB);
        vf.u[1] = *(const uint4*)(pv + kbB + 16);
        o[j] = __builtin_amdgcn_wmma_f32_16x16x32_bf16(
            false, pf.v, false, vf.v, (short)0, o[j], false, false);
      }
    }
  }

  // epilogue: normalize, write to [B,T,H*Dh] bf16 for the Wo GEMM
  const int h = bh % kH, b = bh / kH;
  #pragma unroll
  for (int r = 0; r < 8; ++r) {
    const int q = qbase + wave * 16 + 8 * half + r;
    const float inv = 1.0f / l_r[r];
    #pragma unroll
    for (int j = 0; j < 4; ++j) {
      int dh = j * 16 + l15;
      Oattn[((size_t)(b * kT + q)) * kDIM + h * kDh + dh] = f2bf(o[j][r] * inv);
    }
  }
}

// ---------------------------------------------------------------------------
// Host launch
// ---------------------------------------------------------------------------
extern "C" void kernel_launch(void* const* d_in, const int* in_sizes, int n_in,
                              void* d_out, int out_size, void* d_ws, size_t ws_size,
                              hipStream_t stream) {
  (void)in_sizes; (void)n_in; (void)out_size; (void)ws_size;
  const float* x  = (const float*)d_in[0];
  const float* Wq = (const float*)d_in[1];
  const float* Wk = (const float*)d_in[2];
  const float* Wv = (const float*)d_in[3];
  const float* Wo = (const float*)d_in[4];
  // d_in[5] = mask: causal mask recomputed analytically on device.

  char* ws = (char*)d_ws;
  const size_t SZ_X = (size_t)kMtok * kDIM * 2;   // 8 MB (bf16)
  const size_t SZ_W = (size_t)kDIM * kDIM * 2;    // 2 MB (bf16)
  size_t off = 0;
  unsigned short* xb     = (unsigned short*)(ws + off); off += SZ_X;
  unsigned short* wqb    = (unsigned short*)(ws + off); off += SZ_W;
  unsigned short* wkb    = (unsigned short*)(ws + off); off += SZ_W;
  unsigned short* wvb    = (unsigned short*)(ws + off); off += SZ_W;
  unsigned short* wob    = (unsigned short*)(ws + off); off += SZ_W;
  unsigned short* qproj  = (unsigned short*)(ws + off); off += SZ_X;
  unsigned short* kproj  = (unsigned short*)(ws + off); off += SZ_X;
  unsigned short* vproj  = (unsigned short*)(ws + off); off += SZ_X;
  unsigned short* qh     = (unsigned short*)(ws + off); off += SZ_X;
  unsigned short* kh     = (unsigned short*)(ws + off); off += SZ_X;
  unsigned short* vh     = (unsigned short*)(ws + off); off += SZ_X;
  unsigned short* attnin = (unsigned short*)(ws + off); off += SZ_X;

  // 1) casts
  cast_f32_bf16<<<4096, 256, 0, stream>>>(x,  xb,  kMtok * kDIM);
  cast_f32_bf16<<<1024, 256, 0, stream>>>(Wq, wqb, kDIM * kDIM);
  cast_f32_bf16<<<1024, 256, 0, stream>>>(Wk, wkb, kDIM * kDIM);
  cast_f32_bf16<<<1024, 256, 0, stream>>>(Wv, wvb, kDIM * kDIM);
  cast_f32_bf16<<<1024, 256, 0, stream>>>(Wo, wob, kDIM * kDIM);

  // 2) Q/K/V projections (x @ W^T)
  dim3 gg(kDIM / TN, kMtok / TM);                  // (8, 32)
  gemm_xwT<true><<<gg, 128, 0, stream>>>(xb, wqb, qproj, kMtok, kDIM, kDIM);
  gemm_xwT<true><<<gg, 128, 0, stream>>>(xb, wkb, kproj, kMtok, kDIM, kDIM);
  gemm_xwT<true><<<gg, 128, 0, stream>>>(xb, wvb, vproj, kMtok, kDIM, kDIM);

  // 3) RoPE + head-major repack (q pre-scaled by Dh^-0.5)
  rope_pack<<<kB * kT * kH, 64, 0, stream>>>(qproj, kproj, vproj, qh, kh, vh);

  // 4) causal flash attention
  flash_attn<<<dim3(kT / 64, kB * kH), 128, 0, stream>>>(qh, kh, vh, attnin);

  // 5) output projection -> fp32 d_out
  gemm_xwT<false><<<gg, 128, 0, stream>>>(attnin, wob, (float*)d_out,
                                          kMtok, kDIM, kDIM);
}
